// Model_39676907880799
// MI455X (gfx1250) — matrix operand order repaired and checked by reading, not compiled, not run on hardware
//
#include <hip/hip_runtime.h>

// ---------------------------------------------------------------------------
// Attention forward for MI455X (gfx1250, wave32, WMMA).
//   q    = query @ W_q + b_q                       (bf16 WMMA GEMM)
//   s    = mask==0 ? -1e32 : (q @ k^T) * SCALE     (bf16 WMMA GEMM, async-LDS)
//   attn = softmax(s)                              (row softmax, in place)
//   out  = attn @ value                            (bf16 WMMA GEMM)
// d_out = [ out (B*S*D f32) | attn (B*S*S f32) ]
// ---------------------------------------------------------------------------

typedef __bf16 bf16_t;
typedef __attribute__((ext_vector_type(16))) __bf16 v16bf;
typedef __attribute__((ext_vector_type(8)))  float  v8f;
typedef __attribute__((ext_vector_type(4)))  __bf16 v4bf;

#define B_  8
#define S_  2048
#define D_  1024
#define SCALE_ 0.03125f
#define NEG_  (-1e32f)

#define TM 128          // macro-tile M
#define TN 128          // macro-tile N
#define KC 32           // K chunk per iteration (one wmma K)
#define LDSK (KC + 8)   // +16B pad per row: kills LDS bank conflicts

// gfx1250 async global->LDS copy path (ASYNCcnt-tracked), guarded so the
// source still compiles on toolchains that don't declare the builtins.
#if defined(__has_builtin)
#if __has_builtin(__builtin_amdgcn_global_load_async_to_lds_b128) && \
    __has_builtin(__builtin_amdgcn_s_wait_asynccnt)
#define USE_ASYNC_LDS 1
#endif
#endif
#ifndef USE_ASYNC_LDS
#define USE_ASYNC_LDS 0
#endif

#if USE_ASYNC_LDS
// builtin signature (from hipcc diagnostic): (v4i AS1*, v4i AS3*, Ii, Ii)
typedef __attribute__((__vector_size__(4 * sizeof(int)))) int v4i_t;
typedef __attribute__((address_space(1))) v4i_t g_v4i;   // global int4
typedef __attribute__((address_space(3))) v4i_t l_v4i;   // LDS int4
#endif

// float -> bf16: native convert (backend emits v_cvt pk forms)
__device__ __forceinline__ bf16_t f2bf(float f) { return (bf16_t)f; }

// A fragment, 16x32 bf16 (ISA 7.12.2 16-bit A layout):
// lanes 0-15: elems 0..7 = K0..7,  elems 8..15 = K16..23  (M = lane)
// lanes16-31: elems 0..7 = K8..15, elems 8..15 = K24..31  (M = lane-16)
__device__ __forceinline__ v16bf load_frag_a(const bf16_t* __restrict__ lds,
                                             int m0, int lane) {
  const int lh = lane & 15;
  const int ks = (lane >> 4) << 3;               // 0 or 8
  const bf16_t* p = lds + (m0 + lh) * LDSK;
  union { float4 f[2]; v16bf v; } u;
  u.f[0] = *(const float4*)(p + ks);             // K = ks   .. ks+7
  u.f[1] = *(const float4*)(p + ks + 16);        // K = ks+16.. ks+23
  return u.v;
}

// B fragment, 32x16 bf16: lanes 0-15 elems = K0..7,K16..23 (N = lane); lanes 16-31 K8..15,K24..31
// LDS tile is stored B-transposed: sB[n][k]
__device__ __forceinline__ v16bf load_frag_b(const bf16_t* __restrict__ lds,
                                             int n0, int lane) {
  const int lh = lane & 15;
  const int kb = (lane >> 4) << 3;               // 0 or 8
  const bf16_t* p = lds + (n0 + lh) * LDSK + kb;
  union { float4 f[2]; v16bf v; } u;
  u.f[0] = *(const float4*)(p);
  u.f[1] = *(const float4*)(p + 16);
  return u.v;
}

#define WMMA_BF16(A, Bm, C) \
  __builtin_amdgcn_wmma_f32_16x16x32_bf16(false, (A), false, (Bm), (short)0, (C), false, false)

// ---------------------------------------------------------------------------
// Kernel 1: convert key/value f32 -> bf16 (vectorized x4)
// ---------------------------------------------------------------------------
__global__ __launch_bounds__(256) void cvt_kv_kernel(
    const float* __restrict__ key, const float* __restrict__ value,
    bf16_t* __restrict__ kbf, bf16_t* __restrict__ vbf, long n4) {
  long i = (long)blockIdx.x * blockDim.x + threadIdx.x;
  if (i >= n4) return;
  float4 kf = ((const float4*)key)[i];
  float4 vf = ((const float4*)value)[i];
  v4bf kb, vb;
  kb[0] = f2bf(kf.x); kb[1] = f2bf(kf.y); kb[2] = f2bf(kf.z); kb[3] = f2bf(kf.w);
  vb[0] = f2bf(vf.x); vb[1] = f2bf(vf.y); vb[2] = f2bf(vf.z); vb[3] = f2bf(vf.w);
  ((v4bf*)kbf)[i] = kb;
  ((v4bf*)vbf)[i] = vb;
}

// ---------------------------------------------------------------------------
// Kernel 2: projection GEMM  qbf[M=B*S, D] = bf16(query @ W_q + b_q)
// ---------------------------------------------------------------------------
__global__ __launch_bounds__(256) void proj_kernel(
    const float* __restrict__ query, const float* __restrict__ Wq,
    const float* __restrict__ bq, bf16_t* __restrict__ qbf) {
  __shared__ __align__(16) bf16_t sA[TM * LDSK];
  __shared__ __align__(16) bf16_t sB[TN * LDSK];

  const int tid  = threadIdx.x;
  const int lane = tid & 31;
  const int wave = tid >> 5;
  const int waveM = wave >> 1;                   // 0..3 -> 32 rows each
  const int waveN = wave & 1;                    // 0..1 -> 64 cols each
  const int gm0 = blockIdx.x * TM;
  const int gn0 = blockIdx.y * TN;

  v8f acc[2][4] = {};

  const int am = tid >> 1;                       // 0..127 (A row)
  const int ak = (tid & 1) * 16;                 // 0 / 16 (A k half)
  const int bk = tid >> 3;                       // 0..31  (B k row)
  const int bn = (tid & 7) * 16;                 // B n strip

  for (int k0 = 0; k0 < D_; k0 += KC) {
    // --- A tile: 128x32, f32 -> bf16 ---
    const float4* ga = (const float4*)(query + (size_t)(gm0 + am) * D_ + k0 + ak);
    bf16_t* pa = sA + am * LDSK + ak;
#pragma unroll
    for (int v = 0; v < 4; ++v) {
      float4 f = ga[v];
      pa[4 * v + 0] = f2bf(f.x); pa[4 * v + 1] = f2bf(f.y);
      pa[4 * v + 2] = f2bf(f.z); pa[4 * v + 3] = f2bf(f.w);
    }
    // --- B tile: W_q[k0+bk][gn0+bn..+15] transposed into sB[n][k] ---
    const float4* gb = (const float4*)(Wq + (size_t)(k0 + bk) * D_ + gn0 + bn);
#pragma unroll
    for (int v = 0; v < 4; ++v) {
      float4 f = gb[v];
      sB[(bn + 4 * v + 0) * LDSK + bk] = f2bf(f.x);
      sB[(bn + 4 * v + 1) * LDSK + bk] = f2bf(f.y);
      sB[(bn + 4 * v + 2) * LDSK + bk] = f2bf(f.z);
      sB[(bn + 4 * v + 3) * LDSK + bk] = f2bf(f.w);
    }
    if (k0 + KC < D_) {                          // gfx1250 global_prefetch_b8
      __builtin_prefetch(query + (size_t)(gm0 + am) * D_ + k0 + KC + ak, 0, 1);
      __builtin_prefetch(Wq + (size_t)(k0 + KC + bk) * D_ + gn0 + bn, 0, 1);
    }
    __syncthreads();

    v16bf a0 = load_frag_a(sA, waveM * 32 + 0, lane);
    v16bf a1 = load_frag_a(sA, waveM * 32 + 16, lane);
#pragma unroll
    for (int tn = 0; tn < 4; ++tn) {
      v16bf b = load_frag_b(sB, waveN * 64 + tn * 16, lane);
      acc[0][tn] = WMMA_BF16(a0, b, acc[0][tn]);
      acc[1][tn] = WMMA_BF16(a1, b, acc[1][tn]);
    }
    __syncthreads();
  }

  // epilogue: + bias, -> bf16
  const int rbase = (lane >> 4) << 3;
  const int cl = lane & 15;
#pragma unroll
  for (int tm = 0; tm < 2; ++tm)
#pragma unroll
    for (int tn = 0; tn < 4; ++tn) {
      const int col = gn0 + waveN * 64 + tn * 16 + cl;
      const float bias = bq[col];
#pragma unroll
      for (int r = 0; r < 8; ++r) {
        const int row = gm0 + waveM * 32 + tm * 16 + rbase + r;
        qbf[(size_t)row * D_ + col] = f2bf(acc[tm][tn][r] + bias);
      }
    }
}

// ---------------------------------------------------------------------------
// Kernel 3: scores GEMM  attn[b,q,k] = mask? (qbf[b] @ kbf[b]^T)*SCALE : -1e32
//   Tiles are raw bf16 copies -> double-buffered GLOBAL_LOAD_ASYNC_TO_LDS.
// ---------------------------------------------------------------------------
__global__ __launch_bounds__(256) void scores_kernel(
    const bf16_t* __restrict__ qbf, const bf16_t* __restrict__ kbf,
    const int* __restrict__ mask, float* __restrict__ attn) {
  __shared__ __align__(16) bf16_t sA[2][TM * LDSK];
  __shared__ __align__(16) bf16_t sB[2][TN * LDSK];

  const int tid  = threadIdx.x;
  const int lane = tid & 31;
  const int wave = tid >> 5;
  const int waveM = wave >> 1;
  const int waveN = wave & 1;
  const int b   = blockIdx.z;
  const int gm0 = blockIdx.x * TM;               // query rows
  const int gn0 = blockIdx.y * TN;               // key rows (= score cols)

  const int am = tid >> 1;                       // tile row handled by thread
  const int ak = (tid & 1) * 16;                 // k half (16 bf16 = 32B)

  const bf16_t* gA = qbf + (size_t)b * S_ * D_ + (size_t)(gm0 + am) * D_ + ak;
  const bf16_t* gB = kbf + (size_t)b * S_ * D_ + (size_t)(gn0 + am) * D_ + ak;
  bf16_t* lA0 = &sA[0][am * LDSK + ak];
  bf16_t* lB0 = &sB[0][am * LDSK + ak];
  bf16_t* lA1 = &sA[1][am * LDSK + ak];
  bf16_t* lB1 = &sB[1][am * LDSK + ak];

  v8f acc[2][4] = {};

#if USE_ASYNC_LDS
  // 4 async b128 instructions per wave per tile fill (2xA, 2xB via ioffset 0/16)
#define SC_ISSUE(LA, LB, KK)                                                   \
  do {                                                                         \
    __builtin_amdgcn_global_load_async_to_lds_b128((g_v4i*)(gA + (KK)),        \
                                                   (l_v4i*)(LA), 0, 0);        \
    __builtin_amdgcn_global_load_async_to_lds_b128((g_v4i*)(gA + (KK)),        \
                                                   (l_v4i*)(LA), 16, 0);       \
    __builtin_amdgcn_global_load_async_to_lds_b128((g_v4i*)(gB + (KK)),        \
                                                   (l_v4i*)(LB), 0, 0);        \
    __builtin_amdgcn_global_load_async_to_lds_b128((g_v4i*)(gB + (KK)),        \
                                                   (l_v4i*)(LB), 16, 0);       \
  } while (0)
  SC_ISSUE(lA0, lB0, 0);                         // preload buffer 0
#endif

  int buf = 0;
  for (int k0 = 0; k0 < D_; k0 += KC, buf ^= 1) {
#if USE_ASYNC_LDS
    if (k0 + KC < D_) {                          // issue next tile, then wait
      if (buf == 0) SC_ISSUE(lA1, lB1, k0 + KC); // for current tile only:
      else          SC_ISSUE(lA0, lB0, k0 + KC); // async loads retire in order
      __builtin_amdgcn_s_wait_asynccnt(4);
    } else {
      __builtin_amdgcn_s_wait_asynccnt(0);
    }
#else
    bf16_t* lA = buf ? lA1 : lA0;
    bf16_t* lB = buf ? lB1 : lB0;
    *(float4*)lA = *(const float4*)(gA + k0);
    *(float4*)lB = *(const float4*)(gB + k0);
    if (k0 + KC < D_) {
      __builtin_prefetch(gA + k0 + KC, 0, 1);
      __builtin_prefetch(gB + k0 + KC, 0, 1);
    }
#endif
    __syncthreads();

    const bf16_t* tileA = &sA[buf][0];
    const bf16_t* tileB = &sB[buf][0];
    v16bf a0 = load_frag_a(tileA, waveM * 32 + 0, lane);
    v16bf a1 = load_frag_a(tileA, waveM * 32 + 16, lane);
#pragma unroll
    for (int tn = 0; tn < 4; ++tn) {
      v16bf bm = load_frag_b(tileB, waveN * 64 + tn * 16, lane);
      acc[0][tn] = WMMA_BF16(a0, bm, acc[0][tn]);
      acc[1][tn] = WMMA_BF16(a1, bm, acc[1][tn]);
    }
    __syncthreads();                             // protect buffer reuse
  }

  // epilogue: scale + mask -> pre-softmax scores (f32) into attn region
  const int*  mb = mask + (size_t)b * S_ * S_;
  float*      ab = attn + (size_t)b * S_ * S_;
  const int rbase = (lane >> 4) << 3;
  const int cl = lane & 15;
#pragma unroll
  for (int tm = 0; tm < 2; ++tm)
#pragma unroll
    for (int tn = 0; tn < 4; ++tn) {
      const int col = gn0 + waveN * 64 + tn * 16 + cl;
#pragma unroll
      for (int r = 0; r < 8; ++r) {
        const int row = gm0 + waveM * 32 + tm * 16 + rbase + r;
        const int mv  = mb[(size_t)row * S_ + col];
        ab[(size_t)row * S_ + col] =
            (mv == 0) ? NEG_ : acc[tm][tn][r] * SCALE_;
      }
    }
}

// ---------------------------------------------------------------------------
// Kernel 4: in-place row softmax over attn (B*S rows of S floats)
// ---------------------------------------------------------------------------
__global__ __launch_bounds__(256) void softmax_kernel(float* __restrict__ attn) {
  __shared__ float red[256];
  const int tid = threadIdx.x;
  float* p = attn + (size_t)blockIdx.x * S_;

  float x[8];
  float m = -3.4e38f;
#pragma unroll
  for (int i = 0; i < 8; ++i) { x[i] = p[tid + i * 256]; m = fmaxf(m, x[i]); }
  red[tid] = m; __syncthreads();
  for (int s = 128; s > 0; s >>= 1) {
    if (tid < s) red[tid] = fmaxf(red[tid], red[tid + s]);
    __syncthreads();
  }
  m = red[0]; __syncthreads();

  float sum = 0.f;
#pragma unroll
  for (int i = 0; i < 8; ++i) { x[i] = __expf(x[i] - m); sum += x[i]; }
  red[tid] = sum; __syncthreads();
  for (int s = 128; s > 0; s >>= 1) {
    if (tid < s) red[tid] += red[tid + s];
    __syncthreads();
  }
  const float inv = 1.0f / red[0];
#pragma unroll
  for (int i = 0; i < 8; ++i) p[tid + i * 256] = x[i] * inv;
}

// ---------------------------------------------------------------------------
// Kernel 5: out GEMM  out[b] = attn[b] (f32 -> bf16 on load) @ vbf[b]
// ---------------------------------------------------------------------------
__global__ __launch_bounds__(256) void av_kernel(
    const float* __restrict__ attn, const bf16_t* __restrict__ vbf,
    float* __restrict__ outp) {
  __shared__ __align__(16) bf16_t sA[TM * LDSK];
  __shared__ __align__(16) bf16_t sB[TN * LDSK];

  const int tid  = threadIdx.x;
  const int lane = tid & 31;
  const int wave = tid >> 5;
  const int waveM = wave >> 1;
  const int waveN = wave & 1;
  const int b   = blockIdx.z;
  const int gm0 = blockIdx.x * TM;               // query rows
  const int gn0 = blockIdx.y * TN;               // feature cols

  const float*  ab = attn + (size_t)b * S_ * S_;
  const bf16_t* vb = vbf  + (size_t)b * S_ * D_;

  v8f acc[2][4] = {};

  const int am = tid >> 1;
  const int ak = (tid & 1) * 16;
  const int bk = tid >> 3;
  const int bn = (tid & 7) * 16;

  for (int k0 = 0; k0 < S_; k0 += KC) {
    // A tile: attn rows f32 -> bf16
    const float4* ga = (const float4*)(ab + (size_t)(gm0 + am) * S_ + k0 + ak);
    bf16_t* pa = sA + am * LDSK + ak;
#pragma unroll
    for (int v = 0; v < 4; ++v) {
      float4 f = ga[v];
      pa[4 * v + 0] = f2bf(f.x); pa[4 * v + 1] = f2bf(f.y);
      pa[4 * v + 2] = f2bf(f.z); pa[4 * v + 3] = f2bf(f.w);
    }
    // B tile: value rows [k][n] (bf16) transposed into sB[n][k]
    union { float4 f[2]; v16bf v; } ub;
    const bf16_t* gv = vb + (size_t)(k0 + bk) * D_ + gn0 + bn;
    ub.f[0] = *(const float4*)(gv);
    ub.f[1] = *(const float4*)(gv + 8);
#pragma unroll
    for (int j = 0; j < 16; ++j) sB[(bn + j) * LDSK + bk] = ub.v[j];
    if (k0 + KC < S_) {
      __builtin_prefetch(ab + (size_t)(gm0 + am) * S_ + k0 + KC + ak, 0, 1);
      __builtin_prefetch(vb + (size_t)(k0 + KC + bk) * D_ + gn0 + bn, 0, 1);
    }
    __syncthreads();

    v16bf a0 = load_frag_a(sA, waveM * 32 + 0, lane);
    v16bf a1 = load_frag_a(sA, waveM * 32 + 16, lane);
#pragma unroll
    for (int tn = 0; tn < 4; ++tn) {
      v16bf bm = load_frag_b(sB, waveN * 64 + tn * 16, lane);
      acc[0][tn] = WMMA_BF16(a0, bm, acc[0][tn]);
      acc[1][tn] = WMMA_BF16(a1, bm, acc[1][tn]);
    }
    __syncthreads();
  }

  // epilogue: f32 out
  const int rbase = (lane >> 4) << 3;
  const int cl = lane & 15;
#pragma unroll
  for (int tm = 0; tm < 2; ++tm)
#pragma unroll
    for (int tn = 0; tn < 4; ++tn) {
      const int col = gn0 + waveN * 64 + tn * 16 + cl;
#pragma unroll
      for (int r = 0; r < 8; ++r) {
        const int row = gm0 + waveM * 32 + tm * 16 + rbase + r;
        outp[((size_t)b * S_ + row) * D_ + col] = acc[tm][tn][r];
      }
    }
}

// ---------------------------------------------------------------------------
extern "C" void kernel_launch(void* const* d_in, const int* in_sizes, int n_in,
                              void* d_out, int out_size, void* d_ws, size_t ws_size,
                              hipStream_t stream) {
  const float* query = (const float*)d_in[0];
  const float* key   = (const float*)d_in[1];
  const float* value = (const float*)d_in[2];
  const int*   mask  = (const int*)d_in[3];
  const float* Wq    = (const float*)d_in[4];
  const float* bq    = (const float*)d_in[5];

  float* outp = (float*)d_out;                         // [B,S,D]
  float* attn = outp + (size_t)B_ * S_ * D_;           // [B,S,S]

  const size_t BSD = (size_t)B_ * S_ * D_;
  bf16_t* qbf = (bf16_t*)d_ws;                         // 32 MB
  bf16_t* kbf = qbf + BSD;                             // 32 MB
  bf16_t* vbf = kbf + BSD;                             // 32 MB

  const long n4 = (long)BSD / 4;
  cvt_kv_kernel<<<(int)((n4 + 255) / 256), 256, 0, stream>>>(key, value, kbf, vbf, n4);
  proj_kernel<<<dim3((B_ * S_) / TM, D_ / TN), 256, 0, stream>>>(query, Wq, bq, qbf);
  scores_kernel<<<dim3(S_ / TM, S_ / TN, B_), 256, 0, stream>>>(qbf, kbf, mask, attn);
  softmax_kernel<<<B_ * S_, 256, 0, stream>>>(attn);
  av_kernel<<<dim3(S_ / TM, D_ / TN, B_), 256, 0, stream>>>(attn, vbf, outp);
}
